// DynamicFusionModule_18056042512897
// MI455X (gfx1250) — matrix operand order repaired and checked by reading, not compiled, not run on hardware
//
#include <hip/hip_runtime.h>
#include <math.h>

// ---------------- constants ----------------
#define B_    8
#define C_    256
#define HID_  512
#define NTOK  2304           // 48*48
#define MROWS (B_*NTOK)      // 18432
#define NHEADS 4
#define HD    64
#define QC    128            // query chunk for attention
#define EPSV  1e-5f

// ---------------- vector / frag types ----------------
typedef __attribute__((ext_vector_type(16))) __bf16 v16bf;
typedef __attribute__((ext_vector_type(8)))  float  v8f;
typedef __attribute__((ext_vector_type(8)))  unsigned short us8;

union FragBF { v16bf v; unsigned short u[16]; };

__device__ __forceinline__ unsigned short f2bf16(float f) {
  unsigned int u = __float_as_uint(f);
  unsigned int r = u + 0x7FFFu + ((u >> 16) & 1u);   // round-to-nearest-even
  return (unsigned short)(r >> 16);
}

__device__ __forceinline__ float blockReduceSum(float v, float* sh) {
  int t = threadIdx.x;
  sh[t] = v; __syncthreads();
  #pragma unroll
  for (int s = 128; s > 0; s >>= 1) { if (t < s) sh[t] += sh[t + s]; __syncthreads(); }
  float r = sh[0]; __syncthreads();
  return r;
}

__device__ __forceinline__ float blockReduceMax(float v, float* sh) {
  int t = threadIdx.x;
  sh[t] = v; __syncthreads();
  #pragma unroll
  for (int s = 128; s > 0; s >>= 1) { if (t < s) sh[t] = fmaxf(sh[t], sh[t + s]); __syncthreads(); }
  float r = sh[0]; __syncthreads();
  return r;
}

// ---------------- TDM path selection ----------------
#if defined(__AMDGCN__) && __has_builtin(__builtin_amdgcn_tensor_load_to_lds) && \
    __has_builtin(__builtin_amdgcn_s_wait_tensorcnt)
#define USE_TDM_PATH 1
#else
#define USE_TDM_PATH 0
#endif

#if USE_TDM_PATH
typedef __attribute__((ext_vector_type(4))) unsigned int v4u;
typedef __attribute__((ext_vector_type(8))) int v8i;
typedef __attribute__((ext_vector_type(4))) int v4i;

__device__ __forceinline__ unsigned ldsByteOffset(const void* p) {
  return (unsigned)(unsigned long long)(const __attribute__((address_space(3))) void*)p;
}

// 2D tile load: rows x 32 bf16 elements, row stride ldaElems (elements),
// LDS row stride 48 elements (32 data + 16 pad) via pad_interval=16DW, pad_amount=8DW.
__device__ __forceinline__ void tdm_load_tile(const unsigned short* gptr, unsigned ldsOff,
                                              int ldaElems, int rows) {
  unsigned long long ga = (unsigned long long)gptr;
  v4u g0;
  g0[0] = 1u;                                               // count=1, load, user
  g0[1] = ldsOff;                                           // lds_addr (bytes)
  g0[2] = (unsigned)(ga & 0xFFFFFFFFu);                     // global_addr[31:0]
  g0[3] = (unsigned)((ga >> 32) & 0x01FFFFFFu) | 0x80000000u; // addr[56:32] | type=2
  unsigned td1 = 1u << 20;                                  // generous tensor_dim1 (no OOB)
  v8i g1;
  g1[0] = (int)((1u << 16) | (1u << 20) | (3u << 22) | (7u << 25)); // 2B elems, pad 16DW->+8DW
  g1[1] = (int)(((unsigned)ldaElems & 0xFFFFu) << 16);      // tensor_dim0 lo16 @ [63:48]
  g1[2] = (int)((((unsigned)ldaElems >> 16) & 0xFFFFu) | ((td1 & 0xFFFFu) << 16));
  g1[3] = (int)(((td1 >> 16) & 0xFFFFu) | (32u << 16));     // tile_dim0 = 32 @ [127:112]
  g1[4] = (int)((unsigned)rows & 0xFFFFu);                  // tile_dim1 @ [143:128]
  g1[5] = ldaElems;                                         // tensor_dim0_stride lo32
  g1[6] = 0;
  g1[7] = 0;
  v4i z4 = { 0, 0, 0, 0 };
#if __clang_major__ >= 23
  v8i z8 = {};
  __builtin_amdgcn_tensor_load_to_lds(g0, g1, z4, z4, z8, 0);
#else
  __builtin_amdgcn_tensor_load_to_lds(g0, g1, z4, z4, 0);
#endif
}
#endif  // USE_TDM_PATH

// ---------------- token transpose: [B,C,N] -> [B,N,C] ----------------
__global__ __launch_bounds__(256)
void tok_kernel(const float* __restrict__ fir, const float* __restrict__ fvis,
                float* __restrict__ Xir, float* __restrict__ Xvis) {
  int bn = blockIdx.x, c = threadIdx.x;
  int b = bn / NTOK, n = bn - b * NTOK;
  size_t src = ((size_t)b * C_ + c) * NTOK + n;
  size_t dst = (size_t)bn * C_ + c;
  Xir[dst]  = fir[src];
  Xvis[dst] = fvis[src];
}

// ---------------- sampling agent: per-pixel 512->512(BN,ReLU)->1, sigmoid ----------------
__global__ __launch_bounds__(256)
void agent_kernel(const float* __restrict__ fir, const float* __restrict__ fvis,
                  const float* __restrict__ w1, const float* __restrict__ b1,
                  const float* __restrict__ gam, const float* __restrict__ bet,
                  const float* __restrict__ mean, const float* __restrict__ var,
                  const float* __restrict__ w2, const float* __restrict__ b2,
                  float* __restrict__ wdrv) {
  __shared__ float xsh[2 * C_];
  __shared__ float red[256];
  int bn = blockIdx.x, t = threadIdx.x;
  int b = bn / NTOK, n = bn - b * NTOK;
  for (int c = t; c < 2 * C_; c += 256)
    xsh[c] = (c < C_) ? fir[((size_t)b * C_ + c) * NTOK + n]
                      : fvis[((size_t)b * C_ + (c - C_)) * NTOK + n];
  __syncthreads();
  float acc = 0.f;
  #pragma unroll
  for (int jj = 0; jj < 2; ++jj) {
    int j = t + jj * 256;
    float s = b1[j];
    const float* wr = w1 + (size_t)j * (2 * C_);
    for (int c = 0; c < 2 * C_; ++c) s += wr[c] * xsh[c];
    s = (s - mean[j]) * rsqrtf(var[j] + EPSV) * gam[j] + bet[j];
    s = fmaxf(s, 0.f);
    acc += w2[j] * s;
  }
  float tot = blockReduceSum(acc, red);
  if (t == 0) wdrv[bn] = 1.f / (1.f + expf(-(tot + b2[0])));
}

// ---------------- hypernetwork threshold ----------------
__global__ __launch_bounds__(256)
void hyper_kernel(const float* __restrict__ W, const float* __restrict__ w1,
                  const float* __restrict__ b1, const float* __restrict__ w2,
                  const float* __restrict__ b2, float* __restrict__ T) {
  __shared__ float sh[256];
  int b = blockIdx.x, t = threadIdx.x;
  float s = 0.f;
  for (int n = t; n < NTOK; n += 256) s += W[(size_t)b * NTOK + n];
  s = blockReduceSum(s, sh);
  if (t == 0) {
    float g = s * (1.f / NTOK);
    float acc = b2[0];
    #pragma unroll
    for (int i = 0; i < 16; ++i) acc += w2[i] * fmaxf(w1[i] * g + b1[i], 0.f);
    T[b] = 1.f / (1.f + expf(-acc));
  }
}

// ---------------- soft mask + active set with top-64 fallback ----------------
__global__ __launch_bounds__(256)
void mask_kernel(const float* __restrict__ W, const float* __restrict__ T,
                 float* __restrict__ SOFT, int* __restrict__ ACT) {
  __shared__ int shi[256];
  int b = blockIdx.x, t = threadIdx.x;
  float th = T[b];
  const float* w = W + (size_t)b * NTOK;
  float* sft = SOFT + (size_t)b * NTOK;
  int* act = ACT + (size_t)b * NTOK;
  int cnt = 0;
  for (int n = t; n < NTOK; n += 256) {
    float s = 1.f / (1.f + expf(-(w[n] - th) * 50.f));
    sft[n] = s;
    int a = s > 0.5f;
    act[n] = a; cnt += a;
  }
  shi[t] = cnt; __syncthreads();
  #pragma unroll
  for (int s2 = 128; s2 > 0; s2 >>= 1) { if (t < s2) shi[t] += shi[t + s2]; __syncthreads(); }
  int total = shi[0]; __syncthreads();
  if (total >= 64) return;
  // 64th-largest of positive floats via bit-pattern bisection (exact for distinct values)
  unsigned lo = 0u, hi = 0x3f800000u;
  for (int it = 0; it < 32 && lo < hi; ++it) {
    unsigned mid = lo + ((hi - lo + 1u) >> 1);
    int c = 0;
    for (int n = t; n < NTOK; n += 256) c += (__float_as_uint(w[n]) >= mid);
    shi[t] = c; __syncthreads();
    #pragma unroll
    for (int s2 = 128; s2 > 0; s2 >>= 1) { if (t < s2) shi[t] += shi[t + s2]; __syncthreads(); }
    int cc = shi[0]; __syncthreads();
    if (cc >= 64) lo = mid; else hi = mid - 1u;
  }
  for (int n = t; n < NTOK; n += 256) act[n] = (__float_as_uint(w[n]) >= lo);
}

// ---------------- f32 -> bf16 convert ----------------
__global__ __launch_bounds__(256)
void f32_to_bf16_kernel(const float* __restrict__ in, unsigned short* __restrict__ out, int n) {
  int i = blockIdx.x * 256 + threadIdx.x;
  if (i < n) out[i] = f2bf16(in[i]);
}

// ---------------- layernorm (shared gamma/beta) -> bf16 rows of 256 ----------------
__global__ __launch_bounds__(256)
void ln_bf16_kernel(const float* __restrict__ X, const float* __restrict__ g,
                    const float* __restrict__ bb, unsigned short* __restrict__ out) {
  __shared__ float sh[256];
  size_t row = blockIdx.x; int c = threadIdx.x;
  float x = X[row * C_ + c];
  float mu = blockReduceSum(x, sh) * (1.f / C_);
  float d = x - mu;
  float var = blockReduceSum(d * d, sh) * (1.f / C_);
  out[row * C_ + c] = f2bf16(d * rsqrtf(var + EPSV) * g[c] + bb[c]);
}

// ---------------- generic bf16 WMMA GEMM (NT: out = A[M,K] * W[Nd,K]^T) ----------------
// Block tile 128 x (64*NSUB), BK=32. 8 waves in 2x4: each wave owns 4 M-subtiles x NSUB
// N-subtiles of 16x16 -> 8 (or 4) f32 accumulators per wave. Tiles staged to LDS either
// by the Tensor Data Mover (TENSORcnt) or by per-thread b128 loads.
#define LDT 48   // LDS row stride in bf16 elements (96B: b128-aligned, staggered banks)

template <int NSUB>
__global__ __launch_bounds__(256)
void gemm_bf16_kernel(const unsigned short* __restrict__ A, size_t aStrideZ, int lda,
                      const unsigned short* __restrict__ Bw, size_t bStrideZ, int ldb,
                      const float* __restrict__ bias,
                      int K, int mode,
                      float* __restrict__ outF, unsigned short* __restrict__ outB,
                      size_t outStrideZ, int ldo,
                      unsigned short* __restrict__ Qp, unsigned short* __restrict__ Kp,
                      unsigned short* __restrict__ Vtp, int Ntok, int qc) {
  constexpr int BN = 64 * NSUB;
  __shared__ unsigned short As[128 * LDT];
  __shared__ unsigned short Bs[BN * LDT];

  const int tid  = threadIdx.x;
  const int z    = blockIdx.z;
  const int bm   = blockIdx.y * 128;
  const int bn   = blockIdx.x * BN;
  const int lane = tid & 31;
  const int wave = tid >> 5;
  const int wm   = wave >> 2;       // 0..1 : 64 rows each
  const int wn   = wave & 3;        // 0..3 : 16*NSUB cols each
  const int l15  = lane & 15;
  const int hlf  = lane >> 4;       // lane half 0/1

  const unsigned short* Az = A  + (size_t)z * aStrideZ;
  const unsigned short* Bz = Bw + (size_t)z * bStrideZ;

  v8f acc[4][NSUB];
  #pragma unroll
  for (int s = 0; s < 4; ++s)
    #pragma unroll
    for (int t2 = 0; t2 < NSUB; ++t2) acc[s][t2] = (v8f){};

#if USE_TDM_PATH
  const unsigned asOff = ldsByteOffset(&As[0]);
  const unsigned bsOff = ldsByteOffset(&Bs[0]);
#endif

  for (int k0 = 0; k0 < K; k0 += 32) {
    __syncthreads();                       // everyone done reading previous LDS tiles
#if USE_TDM_PATH
    if (wave == 0) {
      tdm_load_tile(&Az[(size_t)bm * lda + k0], asOff, lda, 128);
      tdm_load_tile(&Bz[(size_t)bn * ldb + k0], bsOff, ldb, BN);
      __builtin_amdgcn_s_wait_tensorcnt(0);
    }
#else
    {
      int idx = tid * 8;
      #pragma unroll
      for (int rep = 0; rep < 2; ++rep, idx += 2048) {     // A: 128x32
        int row = idx >> 5, kc = idx & 31;
        *(us8*)&As[row * LDT + kc] = *(const us8*)&Az[(size_t)(bm + row) * lda + k0 + kc];
      }
      idx = tid * 8;
      #pragma unroll
      for (int rep = 0; rep < NSUB; ++rep, idx += 2048) {  // B: BNx32
        int row = idx >> 5, kc = idx & 31;
        *(us8*)&Bs[row * LDT + kc] = *(const us8*)&Bz[(size_t)(bn + row) * ldb + k0 + kc];
      }
      if (k0 + 32 < K) {
        __builtin_prefetch(&Az[(size_t)(bm + (tid >> 1)) * lda + k0 + 32], 0, 3);
        __builtin_prefetch(&Bz[(size_t)(bn + (tid & 63)) * ldb + k0 + 32], 0, 3);
      }
    }
#endif
    __syncthreads();                       // tiles visible to all waves

    // B fragments: lane = col, contiguous 16 K per lane-half
    FragBF bf[NSUB];
    #pragma unroll
    for (int t2 = 0; t2 < NSUB; ++t2) {
      int bcol = wn * (16 * NSUB) + t2 * 16 + l15;
      int kb = hlf ? 16 : 0;
      us8 p0 = *(const us8*)&Bs[bcol * LDT + kb];
      us8 p1 = *(const us8*)&Bs[bcol * LDT + kb + 8];
      #pragma unroll
      for (int i = 0; i < 8; ++i) { bf[t2].u[i] = p0[i]; bf[t2].u[8 + i] = p1[i]; }
    }
    // A fragments: lane = row, split K halves per lane-half; reuse across NSUB WMMAs
    #pragma unroll
    for (int s = 0; s < 4; ++s) {
      int kb = hlf ? 16 : 0;
      int arow = wm * 64 + s * 16 + l15;
      FragBF af;
      us8 p0 = *(const us8*)&As[arow * LDT + kb];
      us8 p1 = *(const us8*)&As[arow * LDT + kb + 8];
      #pragma unroll
      for (int i = 0; i < 8; ++i) { af.u[i] = p0[i]; af.u[8 + i] = p1[i]; }
      #pragma unroll
      for (int t2 = 0; t2 < NSUB; ++t2)
        acc[s][t2] = __builtin_amdgcn_wmma_f32_16x16x32_bf16(false, af.v, false, bf[t2].v,
                                                             (short)0, acc[s][t2],
                                                             false, false);
    }
  }

  // epilogue: C/D layout -> lane l15 = column, vgpr r -> row r + 8*hlf
  #pragma unroll
  for (int t2 = 0; t2 < NSUB; ++t2) {
    const int gj = bn + wn * (16 * NSUB) + t2 * 16 + l15;
    const float bv = bias ? bias[gj] : 0.f;
    #pragma unroll
    for (int s = 0; s < 4; ++s) {
      #pragma unroll
      for (int r = 0; r < 8; ++r) {
        int gm = bm + wm * 64 + s * 16 + hlf * 8 + r;
        float v = acc[s][t2][r] + bv;
        if (mode == 0) {                       // f32 (+bias)
          outF[(size_t)z * outStrideZ + (size_t)gm * ldo + gj] = v;
        } else if (mode == 1) {                // f32 residual add
          outF[(size_t)gm * ldo + gj] += v;
        } else if (mode == 2) {                // bf16 (+bias)
          outB[(size_t)gm * ldo + gj] = f2bf16(v);
        } else if (mode == 3) {                // bf16 exact-GELU
          float gg = 0.5f * v * (1.f + erff(v * 0.70710678118f));
          outB[(size_t)gm * ldo + gj] = f2bf16(gg);
        } else if (mode == 4) {                // QKV scatter (q scaled by 1/sqrt(hd))
          int part = gj >> 8, hh = (gj >> 6) & 3, dd = gj & 63;
          int bb2 = gm / Ntok, nn = gm - bb2 * Ntok;
          float vv = (part == 0) ? v * 0.125f : v;
          size_t hb = (size_t)(bb2 * NHEADS + hh);
          if (part == 0)      Qp[(hb * Ntok + nn) * HD + dd] = f2bf16(vv);
          else if (part == 1) Kp[(hb * Ntok + nn) * HD + dd] = f2bf16(vv);
          else                Vtp[(hb * HD + dd) * (size_t)Ntok + nn] = f2bf16(vv);
        } else {                               // mode 5: attention-out scatter
          int bb2 = z >> 2, hh = z & 3;
          size_t row = (size_t)bb2 * Ntok + qc + gm;
          outB[row * C_ + hh * HD + gj] = f2bf16(v);
        }
      }
    }
  }
}

// ---------------- masked softmax over keys, writes bf16 probs ----------------
__global__ __launch_bounds__(256)
void softmax_kernel(const float* __restrict__ S, const int* __restrict__ ACT,
                    unsigned short* __restrict__ P) {
  __shared__ float sh[256];
  int bh = blockIdx.y, qi = blockIdx.x, t = threadIdx.x;
  const float* row = S + ((size_t)bh * QC + qi) * NTOK;
  const int* act = ACT + (size_t)(bh >> 2) * NTOK;
  float mx = -3.0e38f;
  for (int n = t; n < NTOK; n += 256) {
    float s = act[n] ? row[n] : -1e9f;
    mx = fmaxf(mx, s);
  }
  mx = blockReduceMax(mx, sh);
  float sum = 0.f;
  for (int n = t; n < NTOK; n += 256) if (act[n]) sum += expf(row[n] - mx);
  sum = blockReduceSum(sum, sh);
  float inv = 1.f / sum;
  unsigned short* prow = P + ((size_t)bh * QC + qi) * NTOK;
  for (int n = t; n < NTOK; n += 256)
    prow[n] = f2bf16(act[n] ? expf(row[n] - mx) * inv : 0.f);
}

// ---------------- final fusion + scatter-back + aux ----------------
__global__ __launch_bounds__(256)
void final_kernel(const float* __restrict__ fir, const float* __restrict__ fvis,
                  const float* __restrict__ Xir, const float* __restrict__ Xvis,
                  const float* __restrict__ W, const float* __restrict__ S,
                  const int* __restrict__ ACT, const float* __restrict__ hs,
                  float* __restrict__ out) {
  int bn = blockIdx.x, c = threadIdx.x;
  int b = bn / NTOK, n = bn - b * NTOK;
  size_t src = ((size_t)b * C_ + c) * NTOK + n;
  float base = fir[src] + fvis[src];
  float val = base;
  if (ACT[bn]) {
    float w = W[bn], s = S[bn];
    size_t p = (size_t)bn * C_ + c;
    float fus = Xir[p] * w + Xvis[p] * (1.f - w);
    val = (base + fus * s) * (1.f + s * hs[0]);
  }
  out[src] = val;
  if (bn == 0 && c == 0) out[(size_t)B_ * C_ * NTOK] = 0.f;   // aux_loss
}

// ================= host orchestration =================
struct MixerW {
  const float *lng, *lnb, *inb, *outb, *f1b, *f2b;
  const unsigned short *inW, *outW, *f1W, *f2W;
};

template <int NSUB>
static void launch_gemm(hipStream_t st, const unsigned short* A, size_t aSZ, int lda,
                        const unsigned short* Bw, size_t bSZ, int ldb, const float* bias,
                        int M, int Nd, int K, int Z, int mode,
                        float* outF, unsigned short* outB, size_t oSZ, int ldo,
                        unsigned short* Qp, unsigned short* Kp, unsigned short* Vtp, int qc) {
  dim3 grid(Nd / (64 * NSUB), M / 128, Z);
  gemm_bf16_kernel<NSUB><<<grid, 256, 0, st>>>(A, aSZ, lda, Bw, bSZ, ldb, bias, K, mode,
                                               outF, outB, oSZ, ldo, Qp, Kp, Vtp, NTOK, qc);
}

static void run_mixer(hipStream_t st, float* X, const MixerW& mw,
                      unsigned short* ABUF, unsigned short* HBUF,
                      unsigned short* Qb, unsigned short* Kb, unsigned short* Vt,
                      float* SCORES, unsigned short* PBF, const int* ACT) {
  // LN1 -> bf16
  ln_bf16_kernel<<<MROWS, 256, 0, st>>>(X, mw.lng, mw.lnb, ABUF);
  // QKV projection with head-layout scatter epilogue
  launch_gemm<2>(st, ABUF, 0, C_, mw.inW, 0, C_, mw.inb, MROWS, 3 * C_, C_, 1, 4,
                 nullptr, nullptr, 0, 0, Qb, Kb, Vt, 0);
  // attention in 128-query chunks, batched over 32 (b,h)
  for (int qc = 0; qc < NTOK; qc += QC) {
    launch_gemm<2>(st, Qb + (size_t)qc * HD, (size_t)NTOK * HD, HD,
                   Kb, (size_t)NTOK * HD, HD, nullptr,
                   QC, NTOK, HD, B_ * NHEADS, 0,
                   SCORES, nullptr, (size_t)QC * NTOK, NTOK, nullptr, nullptr, nullptr, 0);
    softmax_kernel<<<dim3(QC, B_ * NHEADS), 256, 0, st>>>(SCORES, ACT, PBF);
    launch_gemm<1>(st, PBF, (size_t)QC * NTOK, NTOK,
                   Vt, (size_t)HD * NTOK, NTOK, nullptr,
                   QC, HD, NTOK, B_ * NHEADS, 5,
                   nullptr, ABUF, 0, 0, nullptr, nullptr, nullptr, qc);
  }
  // output projection + residual into X
  launch_gemm<2>(st, ABUF, 0, C_, mw.outW, 0, C_, mw.outb, MROWS, C_, C_, 1, 1,
                 X, nullptr, 0, C_, nullptr, nullptr, nullptr, 0);
  // LN2 (shared params) -> bf16
  ln_bf16_kernel<<<MROWS, 256, 0, st>>>(X, mw.lng, mw.lnb, ABUF);
  // FFN: f1 (GELU, bf16 out) then f2 (+residual into X)
  launch_gemm<2>(st, ABUF, 0, C_, mw.f1W, 0, C_, mw.f1b, MROWS, 4 * C_, C_, 1, 3,
                 nullptr, HBUF, 0, 4 * C_, nullptr, nullptr, nullptr, 0);
  launch_gemm<2>(st, HBUF, 0, 4 * C_, mw.f2W, 0, 4 * C_, mw.f2b, MROWS, C_, 4 * C_, 1, 1,
                 X, nullptr, 0, C_, nullptr, nullptr, nullptr, 0);
}

extern "C" void kernel_launch(void* const* d_in, const int* in_sizes, int n_in,
                              void* d_out, int out_size, void* d_ws, size_t ws_size,
                              hipStream_t stream) {
  (void)in_sizes; (void)n_in; (void)out_size; (void)ws_size;
  const float* f_ir  = (const float*)d_in[0];
  const float* f_vis = (const float*)d_in[1];
  const float* aw1 = (const float*)d_in[2];
  const float* ab1 = (const float*)d_in[3];
  const float* bng = (const float*)d_in[4];
  const float* bnb = (const float*)d_in[5];
  const float* bnm = (const float*)d_in[6];
  const float* bnv = (const float*)d_in[7];
  const float* aw2 = (const float*)d_in[8];
  const float* ab2 = (const float*)d_in[9];
  const float* hw1 = (const float*)d_in[10];
  const float* hb1 = (const float*)d_in[11];
  const float* hw2 = (const float*)d_in[12];
  const float* hb2 = (const float*)d_in[13];
  const float* hscale = (const float*)d_in[14];

  char* ws = (char*)d_ws;
  size_t off = 0;
  auto take = [&](size_t bytes) -> char* {
    char* p = ws + off;
    off = (off + bytes + 255) & ~(size_t)255;
    return p;
  };
  float* W_DRV   = (float*)take((size_t)MROWS * 4);
  float* T       = (float*)take((size_t)B_ * 4);
  float* SOFT    = (float*)take((size_t)MROWS * 4);
  int*   ACT     = (int*)  take((size_t)MROWS * 4);
  unsigned short* WBF = (unsigned short*)take((size_t)2 * 786432 * 2);
  float* X_IR    = (float*)take((size_t)MROWS * C_ * 4);
  float* X_VIS   = (float*)take((size_t)MROWS * C_ * 4);
  unsigned short* ABUF = (unsigned short*)take((size_t)MROWS * C_ * 2);
  unsigned short* HBUF = (unsigned short*)take((size_t)MROWS * 4 * C_ * 2);
  unsigned short* Qb   = (unsigned short*)take((size_t)MROWS * C_ * 2);
  unsigned short* Kb   = (unsigned short*)take((size_t)MROWS * C_ * 2);
  unsigned short* Vt   = (unsigned short*)take((size_t)MROWS * C_ * 2);
  float* SCORES  = (float*)take((size_t)B_ * NHEADS * QC * NTOK * 4);
  unsigned short* PBF  = (unsigned short*)take((size_t)B_ * NHEADS * QC * NTOK * 2);

  // tokens
  tok_kernel<<<MROWS, 256, 0, stream>>>(f_ir, f_vis, X_IR, X_VIS);
  // sampling agent / threshold / mask
  agent_kernel<<<MROWS, 256, 0, stream>>>(f_ir, f_vis, aw1, ab1, bng, bnb, bnm, bnv,
                                          aw2, ab2, W_DRV);
  hyper_kernel<<<B_, 256, 0, stream>>>(W_DRV, hw1, hb1, hw2, hb2, T);
  mask_kernel<<<B_, 256, 0, stream>>>(W_DRV, T, SOFT, ACT);

  // convert mixer weights to bf16 (in_w, out_w, f1_w, f2_w) x2
  const int widx[4] = { 17, 19, 21, 23 };
  const int wsz[4]  = { 3 * C_ * C_, C_ * C_, 4 * C_ * C_, C_ * 4 * C_ };
  unsigned short* wptr[8];
  size_t woff = 0;
  for (int m = 0; m < 2; ++m)
    for (int i = 0; i < 4; ++i) {
      const float* src = (const float*)d_in[widx[i] + m * 10];
      wptr[m * 4 + i] = WBF + woff;
      int n = wsz[i];
      f32_to_bf16_kernel<<<(n + 255) / 256, 256, 0, stream>>>(src, WBF + woff, n);
      woff += (size_t)n;
    }

  MixerW mir  = { (const float*)d_in[15], (const float*)d_in[16], (const float*)d_in[18],
                  (const float*)d_in[20], (const float*)d_in[22], (const float*)d_in[24],
                  wptr[0], wptr[1], wptr[2], wptr[3] };
  MixerW mvis = { (const float*)d_in[25], (const float*)d_in[26], (const float*)d_in[28],
                  (const float*)d_in[30], (const float*)d_in[32], (const float*)d_in[34],
                  wptr[4], wptr[5], wptr[6], wptr[7] };

  run_mixer(stream, X_IR,  mir,  ABUF, HBUF, Qb, Kb, Vt, SCORES, PBF, ACT);
  run_mixer(stream, X_VIS, mvis, ABUF, HBUF, Qb, Kb, Vt, SCORES, PBF, ACT);

  final_kernel<<<MROWS, 256, 0, stream>>>(f_ir, f_vis, X_IR, X_VIS, W_DRV, SOFT, ACT,
                                          hscale, (float*)d_out);
}
